// MoEBlock_78288663872291
// MI455X (gfx1250) — matrix operand order repaired and checked by reading, not compiled, not run on hardware
//
#include <hip/hip_runtime.h>
#include <hip/hip_bf16.h>
#include <cstdint>
#include <math.h>

// ---------------------------------------------------------------------------
// MI455X (gfx1250) WMMA transformer-MoE block.
// GEMMs via v_wmma_f32_16x16x32_f16, wave32. Weights pre-swizzled to
// B-fragment order (2x b128 per fragment per lane). Register-double-buffered
// (software pipelined) GEMM inner loops; FFN1 streams W1 panels through the
// Tensor Data Mover into LDS; FFN1/FFN2 use 32x64 wave tiles.
// ---------------------------------------------------------------------------

typedef __attribute__((ext_vector_type(16))) _Float16 v16h;
typedef __attribute__((ext_vector_type(8)))  float    v8f;
typedef __attribute__((ext_vector_type(4)))  unsigned int u32x4;
typedef __attribute__((ext_vector_type(4)))  int i32x4;
typedef __attribute__((ext_vector_type(8)))  int i32x8;

#define WMMA(a, b, c) __builtin_amdgcn_wmma_f32_16x16x32_f16( \
    false, (a), false, (b), (short)0, (c), false, false)

union AFrag { v16h v; _Float16 h[16]; uint32_t u[8]; };
union CFrag { v8f  v; float f[8]; };

__device__ __forceinline__ int lane_id() { return threadIdx.x & 31; }

// A fragment: 16x32 f16, rows at `base`, leading dim lda (elements), K contiguous.
__device__ __forceinline__ v16h load_a_frag(const _Float16* base, int lda) {
  int l = lane_id();
  int m = l & 15;
  int hi = l >> 4;
  AFrag a;
  const uint32_t* p = reinterpret_cast<const uint32_t*>(base + (size_t)m * lda);
#pragma unroll
  for (int v = 0; v < 8; ++v) {
    int ku = v + 8 * hi;  // dword index = k/2
    a.u[v] = p[ku];
  }
  return a.v;
}

// B fragment from swizzled storage: frag = 512 halves laid out [lane][8 dwords].
__device__ __forceinline__ v16h load_b_frag_sw(const _Float16* Bsw, size_t frag) {
  const uint32_t* p =
      reinterpret_cast<const uint32_t*>(Bsw) + frag * 256 + lane_id() * 8;
  AFrag b;
#pragma unroll
  for (int v = 0; v < 8; ++v) b.u[v] = p[v];
  return b.v;
}

// B fragment from an LDS-resident swizzled fragment (ds_load_b128 x2).
__device__ __forceinline__ v16h load_b_frag_lds(const _Float16* p) {
  const uint32_t* q = reinterpret_cast<const uint32_t*>(p) + lane_id() * 8;
  AFrag b;
#pragma unroll
  for (int v = 0; v < 8; ++v) b.u[v] = q[v];
  return b.v;
}

// B fragment where K is contiguous in memory: element (k,n) at B[n*ldn + k].
__device__ __forceinline__ v16h load_b_frag_kcontig(const _Float16* B, int ldn) {
  int l = lane_id();
  int n = l & 15;
  int hi = (l >> 4) * 16;
  AFrag b;
  const uint32_t* p = reinterpret_cast<const uint32_t*>(B + (size_t)n * ldn + hi);
#pragma unroll
  for (int v = 0; v < 8; ++v) b.u[v] = p[v];
  return b.v;
}

__device__ __forceinline__ float gelu_tanh(float x) {
  const float c = 0.7978845608028654f;  // sqrt(2/pi)
  float t = c * (x + 0.044715f * x * x * x);
  return 0.5f * x * (1.0f + tanhf(t));
}

// ---------------------------------------------------------------------------
// Tensor Data Mover: 1D copy of `elems` f16 from global to LDS.
// ---------------------------------------------------------------------------
__device__ __forceinline__ void tdm_load_1d(void* lds_dst, const void* gsrc,
                                            unsigned elems) {
  unsigned long long ga = (unsigned long long)(uintptr_t)gsrc;
  unsigned lds = (unsigned)(uintptr_t)lds_dst;
  u32x4 g0;
  g0[0] = 1u;                                   // count=1 (valid user D#)
  g0[1] = lds;                                  // lds_addr (bytes)
  g0[2] = (unsigned)(ga & 0xffffffffu);         // global_addr[31:0]
  g0[3] = (unsigned)((ga >> 32) & 0x01ffffffu)  // global_addr[56:32]
          | (2u << 30);                         // type=2 ("image")
  i32x8 g1;
  g1[0] = (1 << 16);                            // data_size=1 -> 2 bytes
  g1[1] = (int)(elems << 16);                   // tensor_dim0[15:0] in [31:16]
  g1[2] = (int)((elems >> 16) | (1u << 16));    // dim0 hi | tensor_dim1=1
  g1[3] = (int)((elems & 0xffffu) << 16);       // tile_dim0
  g1[4] = 0;                                    // tile_dim1=0, tile_dim2=0
  g1[5] = (int)elems;                           // tensor_dim0_stride lo32
  g1[6] = 0;
  g1[7] = 0;
  i32x4 z4 = {0, 0, 0, 0};
#if defined(__clang_major__) && (__clang_major__ >= 23)
  i32x8 z8 = {0, 0, 0, 0, 0, 0, 0, 0};
  __builtin_amdgcn_tensor_load_to_lds(g0, g1, z4, z4, z8, 0);
#else
  __builtin_amdgcn_tensor_load_to_lds(g0, g1, z4, z4, 0);
#endif
}

// ---------------------------------------------------------------------------
// Weight conversion into swizzled B-fragment layout.
// transpose==1: src is (N,K) row-major; transpose==0: src is (K,N) row-major.
// ---------------------------------------------------------------------------
__global__ void convert_swizzleB_kernel(const float* __restrict__ src,
                                        _Float16* __restrict__ out,
                                        int N, int K, int transpose) {
  size_t mat = (size_t)K * N;
  size_t base = (size_t)blockIdx.y * mat;
  int o = blockIdx.x * blockDim.x + threadIdx.x;
  if (o >= (int)mat) return;
  int f = o >> 9;
  int r = o & 511;
  int lane = r >> 4;
  int d = r & 15;
  int nfr = N >> 4;
  int ktile = f / nfr, ntile = f - ktile * nfr;
  int n = (ntile << 4) + (lane & 15);
  int k = (ktile << 5) + ((lane >> 4) << 4) + d;
  float v = transpose ? src[base + (size_t)n * K + k]
                      : src[base + (size_t)k * N + n];
  out[base + o] = (_Float16)v;
}

// ---------------------------------------------------------------------------
// V transpose: QKVh (B*512, 1536) f16 -> Vt [b][h][dim(64)][tok(512)] f16
// ---------------------------------------------------------------------------
__global__ void vtrans_kernel(const _Float16* __restrict__ QKV,
                              _Float16* __restrict__ Vt) {
  int i = blockIdx.x * blockDim.x + threadIdx.x;  // over 4*8*64*512
  int tok = i & 511;
  int j = i >> 9;
  int n = j & 63;
  int h = (j >> 6) & 7;
  int b = j >> 9;
  Vt[i] = QKV[(size_t)(b * 512 + tok) * 1536 + 1024 + h * 64 + n];
}

// ---------------------------------------------------------------------------
// LayerNorm (1 wave per token, D=512, 16 elems/lane)
// ---------------------------------------------------------------------------
__device__ __forceinline__ float wave_sum(float s) {
#pragma unroll
  for (int off = 16; off; off >>= 1) s += __shfl_xor(s, off, 32);
  return s;
}

__global__ void ln1_kernel(const float* __restrict__ tokA,
                           const float* __restrict__ tokB,
                           const float* __restrict__ tokC,
                           const float* __restrict__ g,
                           const float* __restrict__ bt,
                           _Float16* __restrict__ X) {
  int wave = threadIdx.x >> 5;
  int l = lane_id();
  int r = blockIdx.x * 8 + wave;  // token row in (B*512)
  int b = r >> 9, n = r & 511;
  const float* src;
  int gi;
  if (n < 128)      { src = tokA + ((size_t)(b * 128 + n)) * 512;         gi = 0; }
  else if (n < 384) { src = tokC + ((size_t)(b * 256 + (n - 128))) * 512; gi = 1; }
  else              { src = tokB + ((size_t)(b * 128 + (n - 384))) * 512; gi = 2; }
  float vals[16], s = 0.f;
#pragma unroll
  for (int i = 0; i < 16; ++i) { vals[i] = src[l * 16 + i]; s += vals[i]; }
  float mean = wave_sum(s) * (1.0f / 512.0f);
  float v2 = 0.f;
#pragma unroll
  for (int i = 0; i < 16; ++i) { float d = vals[i] - mean; v2 += d * d; }
  float rstd = rsqrtf(wave_sum(v2) * (1.0f / 512.0f) + 1e-5f);
#pragma unroll
  for (int i = 0; i < 16; ++i) {
    int c = l * 16 + i;
    X[(size_t)r * 512 + c] =
        (_Float16)((vals[i] - mean) * rstd * g[gi * 512 + c] + bt[gi * 512 + c]);
  }
}

__global__ void ln2_kernel(const float* __restrict__ T,
                           const float* __restrict__ g,
                           const float* __restrict__ bt,
                           _Float16* __restrict__ X) {
  int wave = threadIdx.x >> 5;
  int l = lane_id();
  int r = blockIdx.x * 8 + wave;
  int n = r & 511;
  int gi = (n < 128) ? 0 : (n < 384 ? 1 : 2);
  const float* src = T + (size_t)r * 512;
  float vals[16], s = 0.f;
#pragma unroll
  for (int i = 0; i < 16; ++i) { vals[i] = src[l * 16 + i]; s += vals[i]; }
  float mean = wave_sum(s) * (1.0f / 512.0f);
  float v2 = 0.f;
#pragma unroll
  for (int i = 0; i < 16; ++i) { float d = vals[i] - mean; v2 += d * d; }
  float rstd = rsqrtf(wave_sum(v2) * (1.0f / 512.0f) + 1e-5f);
#pragma unroll
  for (int i = 0; i < 16; ++i) {
    int c = l * 16 + i;
    X[(size_t)r * 512 + c] =
        (_Float16)((vals[i] - mean) * rstd * g[gi * 512 + c] + bt[gi * 512 + c]);
  }
}

// ---------------------------------------------------------------------------
// QKV GEMM, software pipelined: Y(M x N) f16 = X(M x K) f16 @ Bsw + bias
// ---------------------------------------------------------------------------
__global__ void gemm_qkv_kernel(const _Float16* __restrict__ X,
                                const _Float16* __restrict__ Bsw,
                                const float* __restrict__ bias,
                                _Float16* __restrict__ Y,
                                int N, int K) {
  int wave = threadIdx.x >> 5;
  int row0 = blockIdx.y * 64 + wave * 16;
  int col0 = blockIdx.x * 64;
  int nfr = N >> 4;
  v8f acc[4] = {};
  const _Float16* A = X + (size_t)row0 * K;
  v16h a = load_a_frag(A, K);
  v16h b[4];
#pragma unroll
  for (int t = 0; t < 4; ++t) b[t] = load_b_frag_sw(Bsw, (col0 >> 4) + t);
  for (int k0 = 0; k0 < K; k0 += 32) {
    size_t fn = (size_t)((k0 + 32) >> 5) * nfr + (col0 >> 4);
    __builtin_prefetch(Bsw + (fn + nfr) * 512, 0, 3);
    v16h an = load_a_frag(A + k0 + 32, K);
    v16h bn[4];
#pragma unroll
    for (int t = 0; t < 4; ++t) bn[t] = load_b_frag_sw(Bsw, fn + t);
#pragma unroll
    for (int t = 0; t < 4; ++t) acc[t] = WMMA(a, b[t], acc[t]);
    a = an;
#pragma unroll
    for (int t = 0; t < 4; ++t) b[t] = bn[t];
  }
  int l = lane_id(), n = l & 15, mhi = (l >> 4) * 8;
#pragma unroll
  for (int t = 0; t < 4; ++t) {
    CFrag c; c.v = acc[t];
    int col = col0 + 16 * t + n;
    float bv = bias[col];
#pragma unroll
    for (int r = 0; r < 8; ++r)
      Y[(size_t)(row0 + r + mhi) * N + col] = (_Float16)(c.f[r] + bv);
  }
}

// ---------------------------------------------------------------------------
// Attention: per (qblock64, head, batch). Q fragments hoisted; K/V pipelined.
// ---------------------------------------------------------------------------
__global__ void attn_kernel(const _Float16* __restrict__ QKV,
                            const _Float16* __restrict__ Vt,
                            _Float16* __restrict__ O) {
  extern __shared__ char smem[];
  float*    Sall = reinterpret_cast<float*>(smem);
  _Float16* Pall = reinterpret_cast<_Float16*>(smem + 4 * 16 * 512 * sizeof(float));
  int wave = threadIdx.x >> 5;
  int l = lane_id();
  int qb = blockIdx.x, h = blockIdx.y, b = blockIdx.z;
  int rowbase = b * 512;
  int q0 = qb * 64 + wave * 16;  // query row within batch
  float*    S = Sall + wave * 16 * 512;
  _Float16* P = Pall + wave * 16 * 512;
  const _Float16* Q  = QKV + (size_t)(rowbase + q0) * 1536 + h * 64;
  const _Float16* KB = QKV + (size_t)rowbase * 1536 + 512 + h * 64;
  const _Float16* Vh = Vt + ((size_t)(b * 8 + h)) * 64 * 512;
  int qseg = (q0 < 128) ? 0 : (q0 < 384 ? 1 : 2);
  int n = l & 15, mhi = (l >> 4) * 8;

  // Q fragments are invariant over the column loop
  v16h aq0 = load_a_frag(Q, 1536);
  v16h aq1 = load_a_frag(Q + 32, 1536);

  // scores strip 16 x 512 = (Q @ K^T) * 1/8 + mask, pipelined over col tiles
  v16h bk0 = load_b_frag_kcontig(KB, 1536);
  v16h bk1 = load_b_frag_kcontig(KB + 32, 1536);
  for (int c0 = 0; c0 < 512; c0 += 16) {
    const _Float16* KN = KB + (size_t)(c0 + 16) * 1536;
    v16h nb0 = load_b_frag_kcontig(KN, 1536);
    v16h nb1 = load_b_frag_kcontig(KN + 32, 1536);
    v8f acc = {};
    acc = WMMA(aq0, bk0, acc);
    acc = WMMA(aq1, bk1, acc);
    int kseg = (c0 < 128) ? 0 : (c0 < 384 ? 1 : 2);
    float maskadd = (kseg <= qseg) ? 0.0f : -1e9f;
    CFrag c; c.v = acc;
#pragma unroll
    for (int r = 0; r < 8; ++r)
      S[(r + mhi) * 512 + c0 + n] = c.f[r] * 0.125f + maskadd;
    bk0 = nb0; bk1 = nb1;
  }
  __syncthreads();

  // row softmax: lanes 0..15 each own one query row
  if (l < 16) {
    float* row = S + l * 512;
    float mx = -1e30f;
    for (int c = 0; c < 512; ++c) mx = fmaxf(mx, row[c]);
    float sum = 0.f;
    for (int c = 0; c < 512; ++c) { float e = __expf(row[c] - mx); row[c] = e; sum += e; }
    float inv = 1.0f / sum;
    _Float16* prow = P + l * 512;
    for (int c = 0; c < 512; ++c) prow[c] = (_Float16)(row[c] * inv);
  }
  __syncthreads();

  // O strip 16 x 64 = P @ V, V fragments pipelined
  v8f acc[4] = {};
  v16h bv[4];
#pragma unroll
  for (int t = 0; t < 4; ++t)
    bv[t] = load_b_frag_kcontig(Vh + (size_t)(16 * t) * 512, 512);
  for (int k0 = 0; k0 < 512; k0 += 32) {
    v16h bn[4];
#pragma unroll
    for (int t = 0; t < 4; ++t)
      bn[t] = load_b_frag_kcontig(Vh + (size_t)(16 * t) * 512 + k0 + 32, 512);
    v16h a = load_a_frag(P + k0, 512);
#pragma unroll
    for (int t = 0; t < 4; ++t) acc[t] = WMMA(a, bv[t], acc[t]);
#pragma unroll
    for (int t = 0; t < 4; ++t) bv[t] = bn[t];
  }
#pragma unroll
  for (int t = 0; t < 4; ++t) {
    CFrag c; c.v = acc[t];
#pragma unroll
    for (int r = 0; r < 8; ++r)
      O[(size_t)(rowbase + q0 + r + mhi) * 512 + h * 64 + 16 * t + n] =
          (_Float16)c.f[r];
  }
}

// ---------------------------------------------------------------------------
// Output projection + residual, pipelined: T = tokens + O @ Wo^T + bo
// ---------------------------------------------------------------------------
__global__ void gemm_oproj_kernel(const _Float16* __restrict__ O,
                                  const _Float16* __restrict__ Bsw,
                                  const float* __restrict__ bias,
                                  const float* __restrict__ tokA,
                                  const float* __restrict__ tokC,
                                  const float* __restrict__ tokB,
                                  float* __restrict__ T) {
  int wave = threadIdx.x >> 5;
  int row0 = blockIdx.y * 64 + wave * 16;
  int col0 = blockIdx.x * 64;
  const int nfr = 512 >> 4;
  const _Float16* A = O + (size_t)row0 * 512;
  v8f acc[4] = {};
  v16h a = load_a_frag(A, 512);
  v16h b[4];
#pragma unroll
  for (int t = 0; t < 4; ++t) b[t] = load_b_frag_sw(Bsw, (col0 >> 4) + t);
  for (int k0 = 0; k0 < 512; k0 += 32) {
    size_t fn = (size_t)((k0 + 32) >> 5) * nfr + (col0 >> 4);
    v16h an = load_a_frag(A + k0 + 32, 512);
    v16h bn[4];
#pragma unroll
    for (int t = 0; t < 4; ++t) bn[t] = load_b_frag_sw(Bsw, fn + t);
#pragma unroll
    for (int t = 0; t < 4; ++t) acc[t] = WMMA(a, b[t], acc[t]);
    a = an;
#pragma unroll
    for (int t = 0; t < 4; ++t) b[t] = bn[t];
  }
  int l = lane_id(), n = l & 15, mhi = (l >> 4) * 8;
#pragma unroll
  for (int t = 0; t < 4; ++t) {
    CFrag c; c.v = acc[t];
    int col = col0 + 16 * t + n;
    float bv = bias[col];
#pragma unroll
    for (int r = 0; r < 8; ++r) {
      int row = row0 + r + mhi;
      int bb = row >> 9, nn = row & 511;
      const float* res;
      if (nn < 128)      res = tokA + ((size_t)(bb * 128 + nn)) * 512;
      else if (nn < 384) res = tokC + ((size_t)(bb * 256 + (nn - 128))) * 512;
      else               res = tokB + ((size_t)(bb * 128 + (nn - 384))) * 512;
      T[(size_t)row * 512 + col] = res[col] + c.f[r] + bv;
    }
  }
}

// ---------------------------------------------------------------------------
// Router: softmax gates -> top-k renormalized. One thread per token.
// ---------------------------------------------------------------------------
__global__ void router_kernel(const _Float16* __restrict__ Xn,
                              const float* __restrict__ Wr,
                              const float* __restrict__ br,
                              float* __restrict__ gates,
                              int E, int topk, int Mg, int off, int Ng) {
  int i = blockIdx.x * blockDim.x + threadIdx.x;
  if (i >= Mg) return;
  int grow = (i / Ng) * 512 + off + (i % Ng);
  const _Float16* x = Xn + (size_t)grow * 512;
  float logit[8];
  for (int e = 0; e < E; ++e) logit[e] = br[e];
  for (int d = 0; d < 512; ++d) {
    float xv = (float)x[d];
    for (int e = 0; e < E; ++e) logit[e] += xv * Wr[d * E + e];
  }
  float mx = logit[0];
  for (int e = 1; e < E; ++e) mx = fmaxf(mx, logit[e]);
  float sum = 0.f;
  for (int e = 0; e < E; ++e) { logit[e] = __expf(logit[e] - mx); sum += logit[e]; }
  float inv = 1.0f / sum;
  for (int e = 0; e < E; ++e) logit[e] *= inv;
  float g[8];
  for (int e = 0; e < E; ++e) g[e] = 0.f;
  float tsum = 0.f;
  for (int t = 0; t < topk; ++t) {
    int bi = 0; float bv = -1.f;
    for (int e = 0; e < E; ++e)
      if (g[e] == 0.f && logit[e] > bv) { bv = logit[e]; bi = e; }
    g[bi] = bv; tsum += bv;
  }
  float rinv = 1.0f / tsum;
  for (int e = 0; e < E; ++e) gates[i * E + e] = g[e] * rinv;
}

// ---------------------------------------------------------------------------
// FFN1: H[e, li, f] = gate[li,e] * gelu( Xn @ W1_e + b1_e )   (f16)
// grid (F/64, Mg/128, E), wave tile 32x64. W1 panels (4KB/k-step) stream
// through the Tensor Data Mover into a double-buffered LDS panel shared by
// all 4 waves (128 rows of reuse per panel).
// ---------------------------------------------------------------------------
__global__ void ffn1_kernel(const _Float16* __restrict__ Xn,
                            const _Float16* __restrict__ W1sw,
                            const float* __restrict__ b1,
                            const float* __restrict__ gates,
                            _Float16* __restrict__ H,
                            int E, int Mg, int off, int Ng) {
  __shared__ _Float16 panel[2][2048];  // 2 x 4KB swizzled B panel
  int wave = threadIdx.x >> 5;
  int e = blockIdx.z;
  int li0 = blockIdx.y * 128 + wave * 32;
  int col0 = blockIdx.x * 64;
  int grow0 = (li0 / Ng) * 512 + off + (li0 % Ng);  // 32 rows contiguous
  const _Float16* A = Xn + (size_t)grow0 * 512;
  const _Float16* W = W1sw + (size_t)e * 512 * 2048;  // swizzled expert matrix
  const int nfr = 2048 >> 4;

  if (wave == 0)
    tdm_load_1d(panel[0], W + (size_t)(col0 >> 4) * 512, 2048);

  v8f acc[8] = {};
  for (int ki = 0; ki < 16; ++ki) {
    if (wave == 0) __builtin_amdgcn_s_wait_tensorcnt(0);
    __syncthreads();  // panel[ki&1] valid for all waves
    if (ki + 1 < 16 && wave == 0)
      tdm_load_1d(panel[(ki + 1) & 1],
                  W + ((size_t)(ki + 1) * nfr + (col0 >> 4)) * 512, 2048);
    v16h a0 = load_a_frag(A + ki * 32, 512);
    v16h a1 = load_a_frag(A + (size_t)16 * 512 + ki * 32, 512);
    const _Float16* pb = panel[ki & 1];
#pragma unroll
    for (int t = 0; t < 4; ++t) {
      v16h b = load_b_frag_lds(pb + t * 512);
      acc[t]     = WMMA(a0, b, acc[t]);
      acc[4 + t] = WMMA(a1, b, acc[4 + t]);
    }
    __syncthreads();  // reads done before panel[ki&1] is overwritten
  }

  int l = lane_id(), n = l & 15, mhi = (l >> 4) * 8;
#pragma unroll
  for (int half = 0; half < 2; ++half) {
#pragma unroll
    for (int t = 0; t < 4; ++t) {
      CFrag c; c.v = acc[half * 4 + t];
      int col = col0 + 16 * t + n;
      float bvv = b1[e * 2048 + col];
#pragma unroll
      for (int r = 0; r < 8; ++r) {
        int li = li0 + half * 16 + r + mhi;
        float gt = gates[li * E + e];
        H[((size_t)e * Mg + li) * 2048 + col] =
            (_Float16)(gt * gelu_tanh(c.f[r] + bvv));
      }
    }
  }
}

// ---------------------------------------------------------------------------
// FFN2: out = T + sum_e (H_e @ W2_e) + sum_e gate*b2_e.
// grid (512/64, Mg/128), wave tile 32x64, pipelined A/B fragments.
// ---------------------------------------------------------------------------
__global__ void ffn2_kernel(const _Float16* __restrict__ H,
                            const _Float16* __restrict__ W2sw,
                            const float* __restrict__ b2,
                            const float* __restrict__ gates,
                            const float* __restrict__ T,
                            float* __restrict__ Out,
                            int E, int Mg, int off, int Ng) {
  int wave = threadIdx.x >> 5;
  int li0 = blockIdx.y * 128 + wave * 32;
  int col0 = blockIdx.x * 64;
  const int nfr = 512 >> 4;
  v8f acc[8] = {};
  for (int e = 0; e < E; ++e) {
    const _Float16* A = H + ((size_t)e * Mg + li0) * 2048;
    const _Float16* W = W2sw + (size_t)e * 2048 * 512;
    v16h a0 = load_a_frag(A, 2048);
    v16h a1 = load_a_frag(A + (size_t)16 * 2048, 2048);
    v16h b[4];
#pragma unroll
    for (int t = 0; t < 4; ++t) b[t] = load_b_frag_sw(W, (col0 >> 4) + t);
    for (int k0 = 0; k0 < 2048; k0 += 32) {
      size_t fn = (size_t)((k0 + 32) >> 5) * nfr + (col0 >> 4);
      __builtin_prefetch(W + (fn + nfr) * 512, 0, 3);
      v16h a0n = load_a_frag(A + k0 + 32, 2048);
      v16h a1n = load_a_frag(A + (size_t)16 * 2048 + k0 + 32, 2048);
      v16h bn[4];
#pragma unroll
      for (int t = 0; t < 4; ++t) bn[t] = load_b_frag_sw(W, fn + t);
#pragma unroll
      for (int t = 0; t < 4; ++t) {
        acc[t]     = WMMA(a0, b[t], acc[t]);
        acc[4 + t] = WMMA(a1, b[t], acc[4 + t]);
      }
      a0 = a0n; a1 = a1n;
#pragma unroll
      for (int t = 0; t < 4; ++t) b[t] = bn[t];
    }
  }
  int l = lane_id(), n = l & 15, mhi = (l >> 4) * 8;
#pragma unroll
  for (int half = 0; half < 2; ++half) {
#pragma unroll
    for (int t = 0; t < 4; ++t) {
      CFrag c; c.v = acc[half * 4 + t];
      int col = col0 + 16 * t + n;
#pragma unroll
      for (int r = 0; r < 8; ++r) {
        int li = li0 + half * 16 + r + mhi;
        int grow = (li / Ng) * 512 + off + (li % Ng);
        float bb = 0.f;
        for (int e = 0; e < E; ++e) bb += gates[li * E + e] * b2[e * 512 + col];
        Out[(size_t)grow * 512 + col] = T[(size_t)grow * 512 + col] + c.f[r] + bb;
      }
    }
  }
}

// ---------------------------------------------------------------------------
// Host launcher
// ---------------------------------------------------------------------------
extern "C" void kernel_launch(void* const* d_in, const int* in_sizes, int n_in,
                              void* d_out, int out_size, void* d_ws, size_t ws_size,
                              hipStream_t stream) {
  const float* tokA = (const float*)d_in[0];
  const float* tokB = (const float*)d_in[1];
  const float* tokC = (const float*)d_in[2];
  const float* ln1g = (const float*)d_in[3];
  const float* ln1b = (const float*)d_in[4];
  const float* ln2g = (const float*)d_in[5];
  const float* ln2b = (const float*)d_in[6];
  const float* Wqkv = (const float*)d_in[7];
  const float* bqkv = (const float*)d_in[8];
  const float* Wo   = (const float*)d_in[9];
  const float* bo   = (const float*)d_in[10];
  const float* WrA  = (const float*)d_in[11];
  const float* brA  = (const float*)d_in[12];
  const float* W1A  = (const float*)d_in[13];
  const float* b1A  = (const float*)d_in[14];
  const float* W2A  = (const float*)d_in[15];
  const float* b2A  = (const float*)d_in[16];
  const float* WrC  = (const float*)d_in[17];
  const float* brC  = (const float*)d_in[18];
  const float* W1C  = (const float*)d_in[19];
  const float* b1C  = (const float*)d_in[20];
  const float* W2C  = (const float*)d_in[21];
  const float* b2C  = (const float*)d_in[22];
  const float* WrB  = (const float*)d_in[23];
  const float* brB  = (const float*)d_in[24];
  const float* W1B  = (const float*)d_in[25];
  const float* b1B  = (const float*)d_in[26];
  const float* W2B  = (const float*)d_in[27];
  const float* b2B  = (const float*)d_in[28];
  float* Out = (float*)d_out;

  // Workspace carve-up
  char* ws = (char*)d_ws;
  auto alloc = [&](size_t bytes) -> char* {
    char* p = ws;
    ws += (bytes + 255) & ~(size_t)255;
    return p;
  };
  const int M = 4 * 512;  // 2048 token rows
  _Float16* Xh    = (_Float16*)alloc((size_t)M * 512 * 2);
  _Float16* QKVh  = (_Float16*)alloc((size_t)M * 1536 * 2);
  _Float16* Vth   = (_Float16*)alloc((size_t)M * 512 * 2);
  _Float16* Oh    = (_Float16*)alloc((size_t)M * 512 * 2);
  float*    T     = (float*)   alloc((size_t)M * 512 * 4);
  _Float16* Nh    = (_Float16*)alloc((size_t)M * 512 * 2);
  _Float16* Wqkvh = (_Float16*)alloc((size_t)1536 * 512 * 2);
  _Float16* Woh   = (_Float16*)alloc((size_t)512 * 512 * 2);
  _Float16* W1h   = (_Float16*)alloc((size_t)14 * 512 * 2048 * 2);
  _Float16* W2h   = (_Float16*)alloc((size_t)14 * 2048 * 512 * 2);
  float*    gates = (float*)   alloc((size_t)1024 * 6 * 4);
  _Float16* Hbuf  = (_Float16*)alloc((size_t)6 * 1024 * 2048 * 2);
  (void)alloc(1 << 16);  // pad: pipelined loops overread one step past buffers

  const size_t EW = (size_t)512 * 2048;  // one expert's weight element count
  const int mt = (int)(EW + 255) / 256;  // blocks per expert matrix

  // ---- weight conversion into swizzled B layout ----
  convert_swizzleB_kernel<<<dim3((1536 * 512 + 255) / 256, 1), 256, 0, stream>>>(
      Wqkv, Wqkvh, 1536, 512, 1);
  convert_swizzleB_kernel<<<dim3((512 * 512 + 255) / 256, 1), 256, 0, stream>>>(
      Wo, Woh, 512, 512, 1);
  convert_swizzleB_kernel<<<dim3(mt, 4), 256, 0, stream>>>(W1A, W1h + 0 * EW, 2048, 512, 0);
  convert_swizzleB_kernel<<<dim3(mt, 6), 256, 0, stream>>>(W1C, W1h + 4 * EW, 2048, 512, 0);
  convert_swizzleB_kernel<<<dim3(mt, 4), 256, 0, stream>>>(W1B, W1h + 10 * EW, 2048, 512, 0);
  convert_swizzleB_kernel<<<dim3(mt, 4), 256, 0, stream>>>(W2A, W2h + 0 * EW, 512, 2048, 0);
  convert_swizzleB_kernel<<<dim3(mt, 6), 256, 0, stream>>>(W2C, W2h + 4 * EW, 512, 2048, 0);
  convert_swizzleB_kernel<<<dim3(mt, 4), 256, 0, stream>>>(W2B, W2h + 10 * EW, 512, 2048, 0);

  // ---- LN1 -> x (f16) ----
  ln1_kernel<<<M / 8, 256, 0, stream>>>(tokA, tokB, tokC, ln1g, ln1b, Xh);

  // ---- QKV GEMM (2048 x 1536, K=512) ----
  gemm_qkv_kernel<<<dim3(1536 / 64, M / 64), 128, 0, stream>>>(
      Xh, Wqkvh, bqkv, QKVh, 1536, 512);

  // ---- V transpose for P@V ----
  vtrans_kernel<<<(M * 512) / 256, 256, 0, stream>>>(QKVh, Vth);

  // ---- attention (8 qblocks x 8 heads x 4 batches) ----
  size_t attn_lds = 4 * 16 * 512 * (sizeof(float) + sizeof(_Float16));
  attn_kernel<<<dim3(8, 8, 4), 128, attn_lds, stream>>>(QKVh, Vth, Oh);

  // ---- output projection + residual -> T (f32) ----
  gemm_oproj_kernel<<<dim3(512 / 64, M / 64), 128, 0, stream>>>(
      Oh, Woh, bo, tokA, tokC, tokB, T);

  // ---- LN2 -> n (f16) ----
  ln2_kernel<<<M / 8, 256, 0, stream>>>(T, ln2g, ln2b, Nh);

  // ---- MoE groups (sequential on stream; gates/Hbuf reused) ----
  struct Grp {
    int E, topk, Mg, off, Ng;
    const float *Wr, *br, *b1, *b2;
    const _Float16 *W1, *W2;
  };
  Grp grp[3] = {
      {4, 2, 512, 0, 128,    WrA, brA, b1A, b2A, W1h + 0 * EW,  W2h + 0 * EW},
      {6, 1, 1024, 128, 256, WrC, brC, b1C, b2C, W1h + 4 * EW,  W2h + 4 * EW},
      {4, 2, 512, 384, 128,  WrB, brB, b1B, b2B, W1h + 10 * EW, W2h + 10 * EW},
  };
  for (int g = 0; g < 3; ++g) {
    const Grp& G = grp[g];
    router_kernel<<<(G.Mg + 127) / 128, 128, 0, stream>>>(
        Nh, G.Wr, G.br, gates, G.E, G.topk, G.Mg, G.off, G.Ng);
    ffn1_kernel<<<dim3(2048 / 64, G.Mg / 128, G.E), 128, 0, stream>>>(
        Nh, G.W1, G.b1, gates, Hbuf, G.E, G.Mg, G.off, G.Ng);
    ffn2_kernel<<<dim3(512 / 64, G.Mg / 128), 128, 0, stream>>>(
        Hbuf, G.W2, G.b2, gates, T, Out, G.E, G.Mg, G.off, G.Ng);
  }
}
